// HybridSSMGNN_70153995813363
// MI455X (gfx1250) — compile-verified, matrix operand order checked by reading, not run on hardware
//
#include <hip/hip_runtime.h>
#include <hip/hip_bf16.h>

#define B_SZ   16
#define SEQ    2048
#define EMBED  128
#define DSTATE 16
#define HID    256
#define NCLS   10
#define NEDGE  8192
#define LN_EPS 1e-5f

typedef __attribute__((ext_vector_type(16))) __bf16 v16bf;
typedef __attribute__((ext_vector_type(8)))  float  v8f;
typedef unsigned int u32;
typedef __attribute__((ext_vector_type(4))) u32 u32x4;
typedef __attribute__((ext_vector_type(8))) u32 u32x8;
typedef __attribute__((address_space(3))) float lds_f32;

// K index inside a 32-wide K chunk for the 16-bit WMMA A/B operand layout:
// lanes 0-15 hold K {0..7,16..23}, lanes 16-31 hold K {8..15,24..31}.
__device__ __forceinline__ int kpat(int j, int hi) {
  return (j & 7) + ((j >> 3) << 4) + (hi << 3);
}

__device__ __forceinline__ u32 pack16(int a, int b) {
  return (u32)(a & 0xffff) | ((u32)(b & 0xffff) << 16);
}

// LDS byte offset of a __shared__ pointer (ptrtoint of addrspace(3)).
__device__ __forceinline__ u32 lds_offset(float* p) {
  return (u32)(unsigned long long)(lds_f32*)p;
}

// ---------------------------------------------------------------------------
// Kernel 1: embedding + SSM scan + LayerNorm. One block per batch, thread e
// owns embed row e with its 16-wide dstate in registers. Sequential in t;
// cross-thread reductions via wave32 shuffles + LDS partial combine.
// ---------------------------------------------------------------------------
__global__ __launch_bounds__(128)
void ssm_kernel(const int* __restrict__ tokens, const float* __restrict__ emb,
                const float* __restrict__ A_log, const float* __restrict__ B_w,
                const float* __restrict__ C_w, const float* __restrict__ D_skip,
                const float* __restrict__ ln_g, const float* __restrict__ ln_b,
                float* __restrict__ h_out) {
  const int b = blockIdx.x;
  const int e = threadIdx.x;
  const int wave = e >> 5, lane = e & 31;
  __shared__ float s_red[4][16];   // per-wave b_t partials
  __shared__ float s_red2[4][16];  // per-wave h-sum partials
  __shared__ float s_ln[4][2];     // per-wave LN partials

  float aexp[16], bcol[16], ccol[16], h[16];
#pragma unroll
  for (int d = 0; d < 16; ++d) {
    aexp[d] = __expf(-__expf(A_log[e * DSTATE + d]));
    bcol[d] = B_w[d * EMBED + e];       // B_w[d][e]
    ccol[d] = C_w[e * DSTATE + d];      // C_w[e][d]
    h[d] = 0.0f;
  }
  const float dsk = D_skip[e], g = ln_g[e], bln = ln_b[e];
  const int* tk = tokens + b * SEQ;

  for (int t = 0; t < SEQ; ++t) {
    const int tok = tk[t];
    const float x = emb[tok * EMBED + e];

    // b_t[d] = sum_e x[e] * B_w[d][e]
    float p[16];
#pragma unroll
    for (int d = 0; d < 16; ++d) p[d] = x * bcol[d];
#pragma unroll
    for (int off = 16; off >= 1; off >>= 1)
#pragma unroll
      for (int d = 0; d < 16; ++d) p[d] += __shfl_xor(p[d], off, 32);
    if (lane == 0)
#pragma unroll
      for (int d = 0; d < 16; ++d) s_red[wave][d] = p[d];
    __syncthreads();

    // h update + prepare h-mean reduction
#pragma unroll
    for (int d = 0; d < 16; ++d) {
      const float bt = s_red[0][d] + s_red[1][d] + s_red[2][d] + s_red[3][d];
      h[d] = h[d] * aexp[d] + bt;
      p[d] = h[d];
    }
#pragma unroll
    for (int off = 16; off >= 1; off >>= 1)
#pragma unroll
      for (int d = 0; d < 16; ++d) p[d] += __shfl_xor(p[d], off, 32);
    if (lane == 0)
#pragma unroll
      for (int d = 0; d < 16; ++d) s_red2[wave][d] = p[d];
    __syncthreads();

    // y[e] = mean_e(h)[d] @ C_w[e][d] + D_skip*x
    float y = dsk * x;
#pragma unroll
    for (int d = 0; d < 16; ++d) {
      const float hm =
          (s_red2[0][d] + s_red2[1][d] + s_red2[2][d] + s_red2[3][d]) * (1.0f / 128.0f);
      y += hm * ccol[d];
    }

    // LayerNorm over the 128 embed lanes
    float m1 = y, m2 = y * y;
#pragma unroll
    for (int off = 16; off >= 1; off >>= 1) {
      m1 += __shfl_xor(m1, off, 32);
      m2 += __shfl_xor(m2, off, 32);
    }
    if (lane == 0) { s_ln[wave][0] = m1; s_ln[wave][1] = m2; }
    __syncthreads();
    const float mu  = (s_ln[0][0] + s_ln[1][0] + s_ln[2][0] + s_ln[3][0]) * (1.0f / 128.0f);
    const float ms  = (s_ln[0][1] + s_ln[1][1] + s_ln[2][1] + s_ln[3][1]) * (1.0f / 128.0f);
    const float var = ms - mu * mu;
    h_out[((size_t)b * SEQ + t) * EMBED + e] = g * (y - mu) * rsqrtf(var + LN_EPS) + bln;
    __syncthreads();  // protect LDS reuse next step
  }
}

// ---------------------------------------------------------------------------
// Kernel 2: pre-swizzle a (16*NT x ldk) weight matrix into the per-lane bf16
// layout of the WMMA 16x16x32 B operand. out[(nt*nkc+kc)*32+lane][0..15].
// ---------------------------------------------------------------------------
__global__ void swz_w_kernel(const float* __restrict__ W, __bf16* __restrict__ out,
                             int ldk, int nkc) {
  const int nt = blockIdx.x;
  const int kc = threadIdx.x >> 5;
  const int lane = threadIdx.x & 31;
  const int n = lane & 15, hi = lane >> 4;
  const float* wrow = W + (nt * 16 + n) * ldk + kc * 32;
  __bf16* o = out + (((size_t)nt * nkc + kc) * 32 + lane) * 16;
#pragma unroll
  for (int j = 0; j < 16; ++j) o[j] = (__bf16)wrow[kpat(j, hi)];
}

// ---------------------------------------------------------------------------
// Kernel 3: GNN messages. One wave per 16-edge tile. The 16-row h_src gather
// is done by the Tensor Data Mover in gather mode: one tensor_load_to_lds
// with 16-bit row indices pulls 16 rows x 512B into this wave's LDS tile,
// tracked by TENSORcnt. Then A fragments from LDS, 16 HID tiles x 4 WMMAs
// (K=128), atomic scatter to agg per the C/D layout.
// ---------------------------------------------------------------------------
__global__ __launch_bounds__(128)
void msg_kernel(const float* __restrict__ h, const int* __restrict__ edges,
                const __bf16* __restrict__ wsw, float* __restrict__ agg) {
  __shared__ float s_tile[4][16 * EMBED];  // 32 KB
  const int wave = threadIdx.x >> 5, lane = threadIdx.x & 31;
  const int tile = blockIdx.x * 4 + wave;     // B_SZ*NEDGE/16 = 8192 tiles
  const int b = tile >> 9;                    // 512 tiles per batch
  const int tb = (tile & 511) * 16;
  const int ln15 = lane & 15, hi = lane >> 4;
  const int* esrc = edges + b * 2 * NEDGE + tb;
  const int* edst = edges + b * 2 * NEDGE + NEDGE + tb;
  float* st = s_tile[wave];

  // ---- TDM gather-mode descriptor (CDNA5 ISA) ----
  const unsigned long long gaddr =
      (unsigned long long)(uintptr_t)(h + (size_t)b * SEQ * EMBED);
  u32x4 g0;
  g0[0] = 0x80000001u;                         // count=1, gather_mode=1, idx16
  g0[1] = lds_offset(st);                      // lds_addr
  g0[2] = (u32)gaddr;                          // global_addr[31:0]
  g0[3] = ((u32)(gaddr >> 32) & 0x01FFFFFFu) | (2u << 30);  // addr hi | type=2
  u32x8 g1;
  g1[0] = 2u << 16;                            // data_size = 4 bytes
  g1[1] = (u32)EMBED << 16;                    // tensor_dim0 = 128 (lo16)
  g1[2] = (u32)SEQ << 16;                      // dim0 hi=0 | tensor_dim1 lo16
  g1[3] = (u32)EMBED << 16;                    // dim1 hi=0 | tile_dim0 = 128
  g1[4] = 16u;                                 // tile_dim1 = #indices = 16
  g1[5] = (u32)EMBED;                          // tensor_dim0_stride = 128
  g1[6] = 0u;
  g1[7] = 0u;
  int si[16];
#pragma unroll
  for (int r = 0; r < 16; ++r) si[r] = esrc[r];
  u32x4 g2, g3;
  g2[0] = pack16(si[0], si[1]);   g2[1] = pack16(si[2], si[3]);
  g2[2] = pack16(si[4], si[5]);   g2[3] = pack16(si[6], si[7]);
  g3[0] = pack16(si[8], si[9]);   g3[1] = pack16(si[10], si[11]);
  g3[2] = pack16(si[12], si[13]); g3[3] = pack16(si[14], si[15]);

  asm volatile("tensor_load_to_lds %0, %1, %2, %3"
               :: "s"(g0), "s"(g1), "s"(g2), "s"(g3)
               : "memory");
  __builtin_amdgcn_s_wait_tensorcnt(0);  // TDM done; LDS tile valid for this wave

  // A fragments: 16x128 bf16 as 4 K-chunks of 32
  v16bf a[4];
#pragma unroll
  for (int kc = 0; kc < 4; ++kc)
#pragma unroll
    for (int j = 0; j < 16; ++j)
      a[kc][j] = (__bf16)st[ln15 * EMBED + kc * 32 + kpat(j, hi)];

  int dstm[8];
#pragma unroll
  for (int r = 0; r < 8; ++r) dstm[r] = edst[r + hi * 8];

  const v16bf* bw = (const v16bf*)wsw;
  for (int nt = 0; nt < 16; ++nt) {
    v8f c = {};
#pragma unroll
    for (int kc = 0; kc < 4; ++kc) {
      const v16bf bf = bw[(nt * 4 + kc) * 32 + lane];
      c = __builtin_amdgcn_wmma_f32_16x16x32_bf16(false, a[kc], false, bf,
                                                  (short)0, c, false, false);
    }
    float* aggb = agg + (size_t)b * SEQ * HID + nt * 16 + ln15;
#pragma unroll
    for (int r = 0; r < 8; ++r)
      atomicAdd(aggb + (size_t)dstm[r] * HID, c[r]);
  }
}

// ---------------------------------------------------------------------------
// Kernel 4: upd = relu([h | agg] @ Wupd^T + b), fused mean-pool over SEQ.
// One wave per 16-row tile. The contiguous 8KB h tile is staged to LDS with
// async global->LDS B128 transfers (ASYNCcnt); agg half gathered from L2.
// ---------------------------------------------------------------------------
__global__ __launch_bounds__(256)
void upd_kernel(const float* __restrict__ h, const float* __restrict__ agg,
                const __bf16* __restrict__ wsw, const float* __restrict__ bias,
                float* __restrict__ pooled) {
  __shared__ float s_h[8][16 * EMBED];  // 64 KB
  const int wave = threadIdx.x >> 5, lane = threadIdx.x & 31;
  const int tile = blockIdx.x * 8 + wave;  // B_SZ*SEQ/16 = 2048 tiles
  const int row0 = tile * 16;
  const int b = row0 / SEQ;
  const int ln15 = lane & 15, hi = lane >> 4;
  float* sh = s_h[wave];

  // async-stage 16 contiguous rows (8 KB) of h into this wave's LDS tile
  {
    const u32 lbase = lds_offset(sh);
    const float* hrow0 = h + (size_t)row0 * EMBED;
    for (int i = lane; i < 512; i += 32) {  // 512 x 16B
      const u32 loff = lbase + (u32)i * 16u;
      const unsigned long long ga = (unsigned long long)(uintptr_t)(hrow0 + i * 4);
      asm volatile("global_load_async_to_lds_b128 %0, %1, off"
                   :: "v"(loff), "v"(ga) : "memory");
    }
    asm volatile("s_wait_asynccnt 0x0" ::: "memory");
  }

  const size_t rowm = (size_t)(row0 + ln15);  // A row for this lane

  v16bf a[12];
#pragma unroll
  for (int kc = 0; kc < 12; ++kc)
#pragma unroll
    for (int j = 0; j < 16; ++j) {
      const int k = kc * 32 + kpat(j, hi);
      const float f = (k < EMBED) ? sh[ln15 * EMBED + k]
                                  : agg[rowm * HID + (k - EMBED)];
      a[kc][j] = (__bf16)f;
    }

  const v16bf* bw = (const v16bf*)wsw;
  for (int nt = 0; nt < 16; ++nt) {
    v8f c = {};
#pragma unroll
    for (int kc = 0; kc < 12; ++kc) {
      const v16bf bf = bw[(nt * 12 + kc) * 32 + lane];
      c = __builtin_amdgcn_wmma_f32_16x16x32_bf16(false, a[kc], false, bf,
                                                  (short)0, c, false, false);
    }
    const float bv = bias[nt * 16 + ln15];
    float s = 0.0f;
#pragma unroll
    for (int r = 0; r < 8; ++r) s += fmaxf(c[r] + bv, 0.0f);
    atomicAdd(&pooled[b * HID + nt * 16 + ln15], s * (1.0f / SEQ));
  }
}

// ---------------------------------------------------------------------------
// Kernel 5: classifier head, 160 outputs.
// ---------------------------------------------------------------------------
__global__ void cls_kernel(const float* __restrict__ pooled,
                           const float* __restrict__ cw,
                           const float* __restrict__ cb,
                           float* __restrict__ out) {
  const int id = threadIdx.x;
  if (id >= B_SZ * NCLS) return;
  const int b = id / NCLS, c = id % NCLS;
  float s = cb[c];
  for (int k = 0; k < HID; ++k) s += pooled[b * HID + k] * cw[c * HID + k];
  out[id] = s;
}

extern "C" void kernel_launch(void* const* d_in, const int* in_sizes, int n_in,
                              void* d_out, int out_size, void* d_ws, size_t ws_size,
                              hipStream_t stream) {
  (void)in_sizes; (void)n_in; (void)out_size; (void)ws_size;
  const int*   tokens = (const int*)d_in[0];
  /* d_in[1] lengths: unused by reference math */
  const int*   edges  = (const int*)d_in[2];
  const float* emb    = (const float*)d_in[3];
  const float* A_log  = (const float*)d_in[4];
  const float* B_w    = (const float*)d_in[5];
  const float* C_w    = (const float*)d_in[6];
  const float* D_skip = (const float*)d_in[7];
  const float* ln_g   = (const float*)d_in[8];
  const float* ln_b   = (const float*)d_in[9];
  const float* Wmsg   = (const float*)d_in[10];
  const float* Wupd   = (const float*)d_in[11];
  const float* Wupd_b = (const float*)d_in[12];
  const float* cls_w  = (const float*)d_in[13];
  const float* cls_b  = (const float*)d_in[14];
  float* out = (float*)d_out;

  char* ws = (char*)d_ws;
  float*  h       = (float*)(ws);                                   // 16 MB
  float*  agg     = (float*)(ws + ((size_t)16 << 20));              // 32 MB
  float*  pooled  = (float*)(ws + ((size_t)48 << 20));              // 16 KB
  __bf16* wmsg_sw = (__bf16*)(ws + ((size_t)48 << 20) + (1 << 16)); // 64 KB
  __bf16* wupd_sw = (__bf16*)(ws + ((size_t)48 << 20) + (2 << 16)); // 192 KB

  hipMemsetAsync(agg, 0, (size_t)B_SZ * SEQ * HID * sizeof(float), stream);
  hipMemsetAsync(pooled, 0, (size_t)B_SZ * HID * sizeof(float), stream);

  swz_w_kernel<<<16, 4 * 32, 0, stream>>>(Wmsg, wmsg_sw, EMBED, 4);
  swz_w_kernel<<<16, 12 * 32, 0, stream>>>(Wupd, wupd_sw, EMBED + HID, 12);
  ssm_kernel<<<B_SZ, 128, 0, stream>>>(tokens, emb, A_log, B_w, C_w, D_skip,
                                       ln_g, ln_b, h);
  msg_kernel<<<(B_SZ * NEDGE / 16) / 4, 128, 0, stream>>>(h, edges, wmsg_sw, agg);
  upd_kernel<<<(B_SZ * SEQ / 16) / 8, 256, 0, stream>>>(h, agg, wupd_sw, Wupd_b, pooled);
  cls_kernel<<<1, 192, 0, stream>>>(pooled, cls_w, cls_b, out);
}
